// RRCNGatLayer_16123307229935
// MI455X (gfx1250) — matrix operand order repaired and checked by reading, not compiled, not run on hardware
//
#include <hip/hip_runtime.h>
#include <hip/hip_bf16.h>

#define H 96
#define NEG_SLOPE 0.01f

typedef __attribute__((ext_vector_type(16))) __bf16 v16bf;
typedef __attribute__((ext_vector_type(8)))  float  v8f;

union BF16Frag { v16bf v; unsigned short u[16]; };

__device__ __forceinline__ unsigned short f2bf(float f) {
    unsigned u = __float_as_uint(f);
    unsigned r = u + 0x7FFFu + ((u >> 16) & 1u);   // round-to-nearest-even
    return (unsigned short)(r >> 16);
}
__device__ __forceinline__ float bf2f(unsigned short b) {
    return __uint_as_float((unsigned)b << 16);
}
__device__ __forceinline__ int fkey(float f) {     // order-preserving float->int
    int i = __float_as_int(f);
    return i >= 0 ? i : (i ^ 0x7FFFFFFF);
}
__device__ __forceinline__ float fkey_inv(int k) {
    return __int_as_float(k >= 0 ? k : (k ^ 0x7FFFFFFF));
}

// ---------------------------------------------------------------- init fill
__global__ void init_ws_kernel(float* zbase, size_t nZero, int* keys, size_t nKeys) {
    size_t i = (size_t)blockIdx.x * blockDim.x + threadIdx.x;
    size_t stride = (size_t)gridDim.x * blockDim.x;
    for (size_t j = i; j < nZero; j += stride) zbase[j] = 0.0f;
    for (size_t j = i; j < nKeys; j += stride) keys[j] = (int)0x80000000;
}

// ------------------------------------------------- fold/pack weights (tiny)
__global__ void pack_weights_kernel(
    const float* __restrict__ Wm0, const float* __restrict__ Wm1, const float* __restrict__ Wm2,
    const float* __restrict__ bm0, const float* __restrict__ bm1, const float* __restrict__ bm2,
    const float* __restrict__ ef0, const float* __restrict__ ef1, const float* __restrict__ ef2,
    const float* __restrict__ Wn0, const float* __restrict__ Wa0,
    const float* __restrict__ Wn1, const float* __restrict__ Wa1,
    const float* __restrict__ Wu,
    unsigned short* __restrict__ WbigU,   // [96][576] bf16
    unsigned short* __restrict__ WubU,    // [384][96] bf16
    float* __restrict__ V,                // [4][96]  el0,er0,el1,er1
    float* __restrict__ Cc)               // [3][96]  ef@Wm[192:208]+bm
{
    int t = blockIdx.x * blockDim.x + threadIdx.x;
    int stride = gridDim.x * blockDim.x;
    // Wbig columns: [P0 Q0 P1 Q1 P2 Q2]
    for (int i = t; i < 96 * 576; i += stride) {
        int k = i / 576, col = i % 576;
        int type = col / 192, hlf = (col % 192) / 96, j = col % 96;
        const float* Wm = (type == 0) ? Wm0 : (type == 1) ? Wm1 : Wm2;
        WbigU[i] = f2bf(Wm[(hlf * 96 + k) * 96 + j]);
    }
    for (int i = t; i < 384 * 96; i += stride) WubU[i] = f2bf(Wu[i]);
    // attention vectors: V[v][k] = sum_a Wn[k][a] * Wa[v%2 * 32 + a]
    for (int i = t; i < 4 * 96; i += stride) {
        int v = i / 96, k = i % 96;
        const float* Wn = (v >> 1) ? Wn1 : Wn0;
        const float* Wa = (v >> 1) ? Wa1 : Wa0;
        float acc = 0.0f;
        for (int a = 0; a < 32; ++a) acc += Wn[k * 32 + a] * Wa[(v & 1) * 32 + a];
        V[i] = acc;
    }
    // edge constants
    for (int i = t; i < 3 * 96; i += stride) {
        int type = i / 96, j = i % 96;
        const float* Wm = (type == 0) ? Wm0 : (type == 1) ? Wm1 : Wm2;
        const float* ef = (type == 0) ? ef0 : (type == 1) ? ef1 : ef2;
        const float* bm = (type == 0) ? bm0 : (type == 1) ? bm1 : bm2;
        float acc = bm[j];
        for (int d = 0; d < 16; ++d) acc += ef[d] * Wm[(192 + d) * 96 + j];
        Cc[i] = acc;
    }
}

// ---------------------------------- node pre-GEMM: Out = h @ Wbig (bf16 out)
__global__ void node_pregemm_kernel(
    const float* __restrict__ h, const unsigned short* __restrict__ WbigU,
    const float* __restrict__ V, unsigned short* __restrict__ OutB,
    float* __restrict__ att, int N)
{
    __shared__ float          hF[16][96];
    __shared__ unsigned short hB[16][96];
    int node0 = blockIdx.x * 16;
    int t = threadIdx.x;
    for (int i = t; i < 16 * 96; i += 256) {
        int r = i / 96, c = i % 96;
        float v = h[(size_t)(node0 + r) * 96 + c];
        hF[r][c] = v; hB[r][c] = f2bf(v);
    }
    __syncthreads();
    int wave = t >> 5, lane = t & 31;
    int m = lane & 15, hlf = lane >> 4;
    for (int tile = wave; tile < 36; tile += 8) {
        int n0 = tile * 16;
        __builtin_prefetch(WbigU + (size_t)lane * 576 + n0, 0, 3);
        v8f acc = {};
        for (int kk = 0; kk < 3; ++kk) {
            int k0 = kk * 32;
            BF16Frag a, b;
            // A 16x32 bf16 layout (ISA 7.12.2): lane m = row, two 8-elem K chunks
            #pragma unroll
            for (int e = 0; e < 8; ++e) a.u[e]     = hB[m][k0 + hlf * 8 + e];
            #pragma unroll
            for (int e = 0; e < 8; ++e) a.u[8 + e] = hB[m][k0 + 16 + hlf * 8 + e];
            // B 32x16 bf16: lane m = column, two 8-elem K chunks (same as A)
            const unsigned short* bp = WbigU + (size_t)(k0 + hlf * 8) * 576 + n0 + m;
            #pragma unroll
            for (int e = 0; e < 16; ++e) b.u[e] = bp[(size_t)(e < 8 ? e : e + 8) * 576];
            acc = __builtin_amdgcn_wmma_f32_16x16x32_bf16(
                false, a.v, false, b.v, (short)0, acc, false, false);
        }
        #pragma unroll
        for (int r = 0; r < 8; ++r)
            OutB[(size_t)(node0 + r + 8 * hlf) * 576 + n0 + m] = f2bf(acc[r]);
    }
    // per-node attention scalars: el0, er0, el1, er1
    if (t < 64) {
        int v = t >> 4, nd = t & 15;
        const float* vv = V + v * 96;
        float acc = 0.0f;
        for (int k = 0; k < 96; ++k) acc += hF[nd][k] * vv[k];
        att[(size_t)v * N + node0 + nd] = acc;
    }
}

// ----------------------------------- edge scores + segment max (both types)
__global__ void edge_scores_kernel(
    const int* __restrict__ src0, const int* __restrict__ dst0,
    const int* __restrict__ src1, const int* __restrict__ dst1,
    const float* __restrict__ att, float* __restrict__ sbuf,
    int* __restrict__ smaxKey, int N, int E)
{
    int i = blockIdx.x * blockDim.x + threadIdx.x;
    if (i >= 2 * E) return;
    int type = (i >= E);
    int e = type ? (i - E) : i;
    int s = type ? src1[e] : src0[e];
    int d = type ? dst1[e] : dst0[e];
    float x = att[(size_t)(2 * type) * N + s] + att[(size_t)(2 * type + 1) * N + d];
    float sc = x > 0.0f ? x : NEG_SLOPE * x;
    sbuf[i] = sc;
    atomicMax(&smaxKey[(size_t)type * N + d], fkey(sc));
}

// -------------------- attention scatter: exp, den, weighted num (1 wave/edge)
__global__ void attn_scatter_kernel(
    const int* __restrict__ src0, const int* __restrict__ dst0,
    const int* __restrict__ src1, const int* __restrict__ dst1,
    const float* __restrict__ sbuf, const int* __restrict__ smaxKey,
    const unsigned short* __restrict__ OutB, const float* __restrict__ Cc,
    float* __restrict__ num01, float* __restrict__ den, int N, int E)
{
    int gw = (int)(((size_t)blockIdx.x * blockDim.x + threadIdx.x) >> 5);
    int lane = threadIdx.x & 31;
    if (gw >= 2 * E) return;
    int type = (gw >= E);
    int e = type ? (gw - E) : gw;
    int s = type ? src1[e] : src0[e];
    int d = type ? dst1[e] : dst0[e];
    float sc = sbuf[gw];
    float smax = fkey_inv(smaxKey[(size_t)type * N + d]);
    float w = __expf(sc - smax);
    if (lane == 0) atomicAdd(&den[(size_t)type * N + d], w);
    const unsigned short* Ps = OutB + (size_t)s * 576 + type * 192;
    const unsigned short* Qd = OutB + (size_t)d * 576 + type * 192 + 96;
    const float* c  = Cc + type * 96;
    float* nrow = num01 + ((size_t)type * N + d) * 96;
    #pragma unroll
    for (int k = lane; k < 96; k += 32) {
        float mval = bf2f(Ps[k]) + bf2f(Qd[k]) + c[k];
        atomicAdd(&nrow[k], w * mval);
    }
}

// -------------------------------- mean scatter for type 2 (1 wave/edge)
__global__ void mean_scatter_kernel(
    const int* __restrict__ src2, const int* __restrict__ dst2,
    const unsigned short* __restrict__ OutB, const float* __restrict__ Cc,
    float* __restrict__ num2, float* __restrict__ cnt, int N, int E)
{
    int gw = (int)(((size_t)blockIdx.x * blockDim.x + threadIdx.x) >> 5);
    int lane = threadIdx.x & 31;
    if (gw >= E) return;
    int s = src2[gw], d = dst2[gw];
    if (lane == 0) atomicAdd(&cnt[d], 1.0f);
    const unsigned short* Ps = OutB + (size_t)s * 576 + 384;
    const unsigned short* Qd = OutB + (size_t)d * 576 + 480;
    const float* c  = Cc + 192;
    float* nrow = num2 + (size_t)d * 96;
    #pragma unroll
    for (int k = lane; k < 96; k += 32)
        atomicAdd(&nrow[k], bf2f(Ps[k]) + bf2f(Qd[k]) + c[k]);
}

// ------------- final: normalize msgs, concat with h, GEMM @ Wu, bias, ReLU
__global__ void final_gemm_kernel(
    const float* __restrict__ h,
    const float* __restrict__ num0, const float* __restrict__ den0,
    const float* __restrict__ num1, const float* __restrict__ den1,
    const float* __restrict__ num2, const float* __restrict__ cnt,
    const unsigned short* __restrict__ WubU, const float* __restrict__ bu,
    float* __restrict__ out, int N)
{
    __shared__ unsigned short xB[16][384];
    __shared__ float inv[16][3];
    int node0 = blockIdx.x * 16;
    int t = threadIdx.x;
    if (t < 48) {
        int r = t / 3, w = t % 3;
        float dv;
        if (w == 0)      dv = fmaxf(den0[node0 + r], 1e-9f);
        else if (w == 1) dv = fmaxf(den1[node0 + r], 1e-9f);
        else             dv = fmaxf(cnt[node0 + r], 1.0f);
        inv[r][w] = 1.0f / dv;
    }
    __syncthreads();
    for (int i = t; i < 16 * 384; i += 256) {
        int r = i / 384, c = i % 384;
        size_t nd = (size_t)(node0 + r);
        float v;
        if (c < 96)       v = h[nd * 96 + c];
        else if (c < 192) v = num0[nd * 96 + (c - 96)]  * inv[r][0];
        else if (c < 288) v = num1[nd * 96 + (c - 192)] * inv[r][1];
        else              v = num2[nd * 96 + (c - 288)] * inv[r][2];
        xB[r][c] = f2bf(v);
    }
    __syncthreads();
    int wave = t >> 5, lane = t & 31;
    int m = lane & 15, hlf = lane >> 4;
    if (wave < 6) {
        int n0 = wave * 16;
        v8f acc = {};
        for (int kk = 0; kk < 12; ++kk) {
            int k0 = kk * 32;
            BF16Frag a, b;
            #pragma unroll
            for (int e = 0; e < 8; ++e) a.u[e]     = xB[m][k0 + hlf * 8 + e];
            #pragma unroll
            for (int e = 0; e < 8; ++e) a.u[8 + e] = xB[m][k0 + 16 + hlf * 8 + e];
            const unsigned short* bp = WubU + (size_t)(k0 + hlf * 8) * 96 + n0 + m;
            #pragma unroll
            for (int e = 0; e < 16; ++e) b.u[e] = bp[(size_t)(e < 8 ? e : e + 8) * 96];
            acc = __builtin_amdgcn_wmma_f32_16x16x32_bf16(
                false, a.v, false, b.v, (short)0, acc, false, false);
        }
        #pragma unroll
        for (int r = 0; r < 8; ++r) {
            int nd = node0 + r + 8 * hlf;
            float v = acc[r] + bu[n0 + m];
            out[(size_t)nd * 96 + n0 + m] = fmaxf(v, 0.0f);
        }
    }
}

extern "C" void kernel_launch(void* const* d_in, const int* in_sizes, int n_in,
                              void* d_out, int out_size, void* d_ws, size_t ws_size,
                              hipStream_t stream) {
    const float* h   = (const float*)d_in[0];
    const int* src0  = (const int*)d_in[1];  const int* dst0 = (const int*)d_in[2];
    const int* src1  = (const int*)d_in[3];  const int* dst1 = (const int*)d_in[4];
    const int* src2  = (const int*)d_in[5];  const int* dst2 = (const int*)d_in[6];
    const float* Wm0 = (const float*)d_in[7];  const float* bm0 = (const float*)d_in[8];  const float* ef0 = (const float*)d_in[9];
    const float* Wm1 = (const float*)d_in[10]; const float* bm1 = (const float*)d_in[11]; const float* ef1 = (const float*)d_in[12];
    const float* Wm2 = (const float*)d_in[13]; const float* bm2 = (const float*)d_in[14]; const float* ef2 = (const float*)d_in[15];
    const float* Wn0 = (const float*)d_in[16]; const float* Wa0 = (const float*)d_in[17];
    const float* Wn1 = (const float*)d_in[18]; const float* Wa1 = (const float*)d_in[19];
    const float* Wu  = (const float*)d_in[20]; const float* bu  = (const float*)d_in[21];
    float* out = (float*)d_out;

    const int N = in_sizes[0] / H;   // 100000
    const int E = in_sizes[1];       // 500000

    char* ws = (char*)d_ws;
    size_t off = 0;
    auto alloc = [&](size_t bytes) -> void* {
        void* p = ws + off;
        off = (off + bytes + 255) & ~(size_t)255;
        return p;
    };
    unsigned short* WbigU = (unsigned short*)alloc((size_t)96 * 576 * 2);
    unsigned short* WubU  = (unsigned short*)alloc((size_t)384 * 96 * 2);
    float* V   = (float*)alloc((size_t)4 * 96 * 4);
    float* Cc  = (float*)alloc((size_t)3 * 96 * 4);
    unsigned short* OutB = (unsigned short*)alloc((size_t)N * 576 * 2);  // bf16 P/Q
    float* att = (float*)alloc((size_t)4 * N * 4);
    float* sbuf = (float*)alloc((size_t)2 * E * 4);
    int* smaxKey = (int*)alloc((size_t)2 * N * 4);
    // contiguous zero-init region: den(2N) | num01(2*N*96) | num2(N*96) | cnt(N)
    size_t nZero = (size_t)2 * N + (size_t)2 * N * 96 + (size_t)N * 96 + (size_t)N;
    float* zbase = (float*)alloc(nZero * 4);
    float* den   = zbase;
    float* num01 = den + (size_t)2 * N;
    float* num2  = num01 + (size_t)2 * N * 96;
    float* cnt   = num2 + (size_t)N * 96;

    init_ws_kernel<<<2048, 256, 0, stream>>>(zbase, nZero, smaxKey, (size_t)2 * N);
    pack_weights_kernel<<<64, 256, 0, stream>>>(
        Wm0, Wm1, Wm2, bm0, bm1, bm2, ef0, ef1, ef2,
        Wn0, Wa0, Wn1, Wa1, Wu, WbigU, WubU, V, Cc);
    node_pregemm_kernel<<<N / 16, 256, 0, stream>>>(h, WbigU, V, OutB, att, N);
    edge_scores_kernel<<<(2 * E + 255) / 256, 256, 0, stream>>>(
        src0, dst0, src1, dst1, att, sbuf, smaxKey, N, E);
    attn_scatter_kernel<<<(2 * E + 7) / 8, 256, 0, stream>>>(
        src0, dst0, src1, dst1, sbuf, smaxKey, OutB, Cc, num01, den, N, E);
    mean_scatter_kernel<<<(E + 7) / 8, 256, 0, stream>>>(
        src2, dst2, OutB, Cc, num2, cnt, N, E);
    final_gemm_kernel<<<N / 16, 256, 0, stream>>>(
        h, num01, den, num01 + (size_t)N * 96, den + N, num2, cnt,
        WubU, bu, out, N);
}
